// PretrainVideoMamba_53266184405748
// MI455X (gfx1250) — compile-verified, matrix operand order checked by reading, not run on hardware
//
#include <hip/hip_runtime.h>

// ---------------- model constants ----------------
#define L_TOK   1568          // FRAMES * NTOK = 8*196
#define BT      3136          // BATCH * L_TOK
#define BT_PAD  3200          // 25 * 128 (padded rows for unguarded A-tile loads)
#define DM      384           // d_model
#define DI      768           // d_inner
#define NCH     1536          // 2*d_inner
#define NK      56            // dt_rank + 2*d_state
#define DTR     24
#define DST     16
#define NB      2             // batch

typedef __attribute__((ext_vector_type(16))) __bf16 v16bf;
typedef __attribute__((ext_vector_type(8)))  float  v8f;

union Frag { v16bf v; uint4 q[2]; unsigned u[8]; };

__device__ inline unsigned pack2bf(float a, float b) {
    union { __bf16 h[2]; unsigned u; } x;
    x.h[0] = (__bf16)a; x.h[1] = (__bf16)b; return x.u;
}
__device__ inline float silu_f(float x) { return x * (1.f / (1.f + __expf(-x))); }

// ---------------- WMMA GEMM: C[M,N] = A[M,K](bf16) * W[N,K](f32->bf16)^T ----------------
// mode 0: C row-major  C[m*N + n]
// mode 1: channel-major: m=(b,l) -> C[(b*N + n)*L_TOK + l]   (LDS-transposed stores)
#define GBM 128
#define GBN 64
#define GBK 64
#define ROWU 36   // 32 data uints (64 bf16) + 4 pad -> 144B rows, 16B aligned chunks

struct alignas(16) StageS { unsigned A[2][GBM * ROWU]; unsigned B[2][GBN * ROWU]; };
union alignas(16) ShU { StageS s; float tr[8][32 * 34]; };

__global__ __launch_bounds__(256)
void wmma_gemm(const __bf16* __restrict__ A, const float* __restrict__ W,
               float* __restrict__ C, int M, int N, int K, int mode)
{
    __shared__ ShU sh;

    const int tid  = threadIdx.x;
    const int lane = tid & 31;
    const int wid  = tid >> 5;
    const int wm   = wid & 3;     // 4 wave-rows of 32
    const int wn   = wid >> 2;    // 2 wave-cols of 32
    const int m0   = blockIdx.y * GBM;
    const int n0   = blockIdx.x * GBN;
    const int la   = lane & 15;
    const int hi4  = (lane >= 16) ? 1 : 0;   // uint4 index of K-subchunk
    const int NKT  = K / GBK;

    // per-thread A staging coordinates (4 x 16B async chunks)
    unsigned ldsA[2][4];
    unsigned long long gA[4];
    #pragma unroll
    for (int c = 0; c < 4; c++) {
        int chunk = tid + c * 256;
        int row = chunk >> 3;
        int kc  = (chunk & 7) * 8;
        #pragma unroll
        for (int bi = 0; bi < 2; bi++)
            ldsA[bi][c] = (unsigned)(size_t)&sh.s.A[bi][row * ROWU + (kc >> 1)];
        gA[c] = (unsigned long long)(size_t)(A + (size_t)(m0 + row) * K + kc);
    }

    float4 rw[4];
    auto loadW = [&](int kt) {
        #pragma unroll
        for (int c = 0; c < 4; c++) {
            int chunk = tid + c * 256;
            int row = chunk >> 4;
            int kc  = (chunk & 15) * 4;
            rw[c] = *(const float4*)(W + (size_t)(n0 + row) * K + kt * GBK + kc);
        }
    };
    auto storeW = [&](int bi) {
        #pragma unroll
        for (int c = 0; c < 4; c++) {
            int chunk = tid + c * 256;
            int row = chunk >> 4;
            int kc  = (chunk & 15) * 4;
            uint2* dst = (uint2*)&sh.s.B[bi][row * ROWU + (kc >> 1)];
            uint2 v; v.x = pack2bf(rw[c].x, rw[c].y); v.y = pack2bf(rw[c].z, rw[c].w);
            *dst = v;
        }
    };
    auto asyncA = [&](int bi, int kt) {
        #pragma unroll
        for (int c = 0; c < 4; c++) {
            unsigned long long ga = gA[c] + (unsigned long long)kt * (GBK * 2);
            // CDNA5 async memory->LDS copy, 16B per lane, tracked by ASYNCcnt
            asm volatile("global_load_async_to_lds_b128 %0, %1, off"
                         :: "v"(ldsA[bi][c]), "v"(ga) : "memory");
        }
    };

    v8f zero = {0.f,0.f,0.f,0.f,0.f,0.f,0.f,0.f};
    v8f acc[2][2];
    #pragma unroll
    for (int i = 0; i < 2; i++)
        #pragma unroll
        for (int j = 0; j < 2; j++) acc[i][j] = zero;

    // prologue: stage tile 0 into buffer 0
    asyncA(0, 0);
    loadW(0);
    storeW(0);
    asm volatile("s_wait_asynccnt 0x0" ::: "memory");
    __syncthreads();

    for (int kt = 0; kt < NKT; kt++) {
        int cur = kt & 1, nxt = cur ^ 1;
        if (kt + 1 < NKT) {
            asyncA(nxt, kt + 1);     // overlap next A tile with compute
            loadW(kt + 1);           // next W tile into registers
        }

        const unsigned* pA = sh.s.A[cur];
        const unsigned* pB = sh.s.B[cur];
        #pragma unroll
        for (int ks = 0; ks < 2; ks++) {
            Frag a[2], b[2];
            #pragma unroll
            for (int i = 0; i < 2; i++) {
                const uint4* p = (const uint4*)&pA[(wm * 32 + i * 16 + la) * ROWU + ks * 16];
                a[i].q[0] = p[hi4];
                a[i].q[1] = p[2 + hi4];
            }
            #pragma unroll
            for (int j = 0; j < 2; j++) {
                const uint4* p = (const uint4*)&pB[(wn * 32 + j * 16 + la) * ROWU + ks * 16];
                b[j].q[0] = p[hi4];
                b[j].q[1] = p[2 + hi4];
            }
            #pragma unroll
            for (int i = 0; i < 2; i++)
                #pragma unroll
                for (int j = 0; j < 2; j++)
                    acc[i][j] = __builtin_amdgcn_wmma_f32_16x16x32_bf16(
                        false, a[i].v, false, b[j].v, (short)0, acc[i][j], false, false);
        }

        if (kt + 1 < NKT) storeW(nxt);
        asm volatile("s_wait_asynccnt 0x0" ::: "memory");
        __syncthreads();
    }

    const int rowadd = (lane >= 16) ? 8 : 0;
    if (mode == 0) {
        #pragma unroll
        for (int i = 0; i < 2; i++) {
            #pragma unroll
            for (int j = 0; j < 2; j++) {
                int mbase = m0 + wm * 32 + i * 16 + rowadd;
                int nn    = n0 + wn * 32 + j * 16 + la;
                #pragma unroll
                for (int r = 0; r < 8; r++) {
                    int mm = mbase + r;
                    if (mm < M) C[(size_t)mm * N + nn] = acc[i][j][r];
                }
            }
        }
    } else {
        // wave-private LDS transpose -> coalesced channel-major stores along l
        __syncthreads();                      // staging LDS is now free; alias as tr
        float* patch = sh.tr[wid];            // 32 x 32, stride 34
        #pragma unroll
        for (int i = 0; i < 2; i++)
            #pragma unroll
            for (int j = 0; j < 2; j++) {
                int rl = i * 16 + rowadd;
                int cl = j * 16 + la;
                #pragma unroll
                for (int r = 0; r < 8; r++)
                    patch[(rl + r) * 34 + cl] = acc[i][j][r];
            }
        int mw = m0 + wm * 32;                // 32-aligned; never straddles batch edge
        if (mw < M) {
            int bb = (mw >= L_TOK) ? 1 : 0;
            int lbase = mw - bb * L_TOK;
            #pragma unroll 4
            for (int c = 0; c < 32; c++) {
                float v = patch[lane * 34 + c];
                C[((size_t)bb * N + n0 + wn * 32 + c) * L_TOK + lbase + lane] = v;
            }
        }
    }
}

// ---------------- patch gather: x -> bf16 A[3136,768] ----------------
__global__ void patch_gather(const float* __restrict__ x, __bf16* __restrict__ Ap)
{
    int m = blockIdx.x;
    int b = m / L_TOK;
    int rem = m - b * L_TOK;
    int t = rem / 196;
    int g = rem - t * 196;
    int gy = g / 14, gx = g - gy * 14;
    for (int k = threadIdx.x; k < 768; k += 256) {
        int c  = k >> 8;
        int py = (k >> 4) & 15;
        int px = k & 15;
        size_t xi = ((((size_t)b * 3 + c) * 8 + t) * 224 + (gy * 16 + py)) * 224 + (gx * 16 + px);
        Ap[(size_t)m * 768 + k] = (__bf16)x[xi];
    }
}

// ---------------- embed epilogue: bias + pos + temporal, zero residual ----------------
__global__ void embed_finish(float* __restrict__ hid, float* __restrict__ res,
                             const float* __restrict__ pb, const float* __restrict__ pos,
                             const float* __restrict__ tpos)
{
    int idx = blockIdx.x * 256 + threadIdx.x;     // BT*DM exact
    int m = idx / DM, n = idx - m * DM;
    int rem = m % L_TOK;
    int t = rem / 196, p = rem - t * 196;
    hid[idx] += pb[n] + pos[p * DM + n] + tpos[t * DM + n];
    res[idx] = 0.f;
}

// ---------------- residual add + RMSNorm -> bf16 for GEMM ----------------
__global__ void rms_in(float* __restrict__ res, const float* __restrict__ hid,
                       const float* __restrict__ nw, __bf16* __restrict__ Xn)
{
    int m = blockIdx.x, tid = threadIdx.x;
    float v[3]; float ss = 0.f;
    #pragma unroll
    for (int i = 0; i < 3; i++) {
        int d = tid + i * 128;
        float r = res[(size_t)m * DM + d] + hid[(size_t)m * DM + d];
        res[(size_t)m * DM + d] = r;
        v[i] = r; ss += r * r;
    }
    for (int off = 16; off; off >>= 1) ss += __shfl_down(ss, off, 32);
    __shared__ float sred[4];
    if ((tid & 31) == 0) sred[tid >> 5] = ss;
    __syncthreads();
    float scale = rsqrtf((sred[0] + sred[1] + sred[2] + sred[3]) / (float)DM + 1e-5f);
    #pragma unroll
    for (int i = 0; i < 3; i++) {
        int d = tid + i * 128;
        Xn[(size_t)m * DM + d] = (__bf16)(v[i] * scale * nw[d]);
    }
}

// ---------------- final RMSNorm -> fp32 output ----------------
__global__ void rms_final(const float* __restrict__ res, const float* __restrict__ hid,
                          const float* __restrict__ nf, float* __restrict__ out)
{
    int m = blockIdx.x, tid = threadIdx.x;
    float v[3]; float ss = 0.f;
    #pragma unroll
    for (int i = 0; i < 3; i++) {
        int d = tid + i * 128;
        float r = res[(size_t)m * DM + d] + hid[(size_t)m * DM + d];
        v[i] = r; ss += r * r;
    }
    for (int off = 16; off; off >>= 1) ss += __shfl_down(ss, off, 32);
    __shared__ float sred[4];
    if ((tid & 31) == 0) sred[tid >> 5] = ss;
    __syncthreads();
    float scale = rsqrtf((sred[0] + sred[1] + sred[2] + sred[3]) / (float)DM + 1e-5f);
    #pragma unroll
    for (int i = 0; i < 3; i++) {
        int d = tid + i * 128;
        out[(size_t)m * DM + d] = v[i] * scale * nf[d];
    }
}

// ---------------- causal depthwise conv (K=4) + SiLU; dir=blockIdx.y flips input ----------------
__global__ void conv_silu(const float* __restrict__ xz,
                          const float* __restrict__ cwf, const float* __restrict__ cbf,
                          const float* __restrict__ cwb, const float* __restrict__ cbb,
                          float* __restrict__ xc)
{
    int dir = blockIdx.y;
    size_t idx = (size_t)blockIdx.x * 256 + threadIdx.x;   // NB*DI*L_TOK exact
    int l = (int)(idx % L_TOK);
    int d = (int)((idx / L_TOK) % DI);
    int b = (int)(idx / ((size_t)L_TOK * DI));
    const float* w    = dir ? cwb : cwf;
    const float* bias = dir ? cbb : cbf;
    const float* xi   = xz + ((size_t)b * NCH + d) * L_TOK;
    float acc = bias[d];
    #pragma unroll
    for (int k = 0; k < 4; k++) {
        int pos = l - 3 + k;
        if (pos >= 0) {
            int src = dir ? (L_TOK - 1 - pos) : pos;
            acc += w[d * 4 + k] * xi[src];
        }
    }
    xc[(((size_t)dir * NB + b) * DI + d) * L_TOK + l] = silu_f(acc);
}

// ---------------- x_proj: xdbl[b,k,l] = sum_d xc[b,d,l] * W[k,d] ----------------
__global__ void xproj_k(const float* __restrict__ xc,
                        const float* __restrict__ xpwf, const float* __restrict__ xpwb,
                        float* __restrict__ xdbl)
{
    int dir = blockIdx.y;
    size_t idx = (size_t)blockIdx.x * 256 + threadIdx.x;   // NB*NK*L_TOK exact
    int l = (int)(idx % L_TOK);
    int k = (int)((idx / L_TOK) % NK);
    int b = (int)(idx / ((size_t)L_TOK * NK));
    const float* w   = (dir ? xpwb : xpwf) + (size_t)k * DI;
    const float* xcb = xc + (((size_t)dir * NB + b) * DI) * L_TOK + l;
    float acc = 0.f;
    for (int d = 0; d < DI; d++) acc += xcb[(size_t)d * L_TOK] * w[d];
    xdbl[(((size_t)dir * NB + b) * NK + k) * L_TOK + l] = acc;
}

// ---------------- dt = softplus(xdbl[:,:24] @ dt_w^T + dt_b) ----------------
__global__ void dt_k(const float* __restrict__ xdbl,
                     const float* __restrict__ dtwf, const float* __restrict__ dtbf,
                     const float* __restrict__ dtwb, const float* __restrict__ dtbb,
                     float* __restrict__ dtout)
{
    int dir = blockIdx.y;
    size_t idx = (size_t)blockIdx.x * 256 + threadIdx.x;   // NB*DI*L_TOK exact
    int l = (int)(idx % L_TOK);
    int d = (int)((idx / L_TOK) % DI);
    int b = (int)(idx / ((size_t)L_TOK * DI));
    const float* w  = (dir ? dtwb : dtwf) + (size_t)d * DTR;
    const float* xd = xdbl + (((size_t)dir * NB + b) * NK) * L_TOK + l;
    float acc = (dir ? dtbb : dtbf)[d];
    #pragma unroll
    for (int r = 0; r < DTR; r++) acc += xd[(size_t)r * L_TOK] * w[r];
    float sp = (acc > 20.f) ? acc : log1pf(__expf(acc));
    dtout[(((size_t)dir * NB + b) * DI + d) * L_TOK + l] = sp;
}

// ---------------- selective scan: thread=(d,n); 16-lane reduce over state ----------------
__global__ void scan_k(const float* __restrict__ dtb, const float* __restrict__ xc,
                       const float* __restrict__ xdbl, const float* __restrict__ xz,
                       const float* __restrict__ Alf, const float* __restrict__ Dpf,
                       const float* __restrict__ Alb, const float* __restrict__ Dpb,
                       float* __restrict__ y)
{
    int b = blockIdx.y, dir = blockIdx.z;
    int d = blockIdx.x * 16 + (threadIdx.x >> 4);
    int n = threadIdx.x & 15;
    const float* Al = dir ? Alb : Alf;
    const float* Dp = dir ? Dpb : Dpf;
    float Aval = -__expf(Al[d * DST + n]);
    float Dv   = Dp[d];
    size_t chan = ((size_t)dir * NB + b) * DI + d;
    const float* dtp = dtb + chan * L_TOK;
    const float* xcp = xc  + chan * L_TOK;
    const float* Bp  = xdbl + ((((size_t)dir * NB + b) * NK) + DTR + n) * L_TOK;
    const float* Cp  = Bp + (size_t)DST * L_TOK;
    const float* zp  = xz + ((size_t)b * NCH + DI + d) * L_TOK;
    float* yp = y + chan * L_TOK;
    float h = 0.f;
    for (int l = 0; l < L_TOK; l++) {
        float dtv = dtp[l];
        float xcv = xcp[l];
        float da  = __expf(dtv * Aval);
        h = da * h + dtv * xcv * Bp[l];
        float yv = h * Cp[l];
        yv += __shfl_down(yv, 8, 16);
        yv += __shfl_down(yv, 4, 16);
        yv += __shfl_down(yv, 2, 16);
        yv += __shfl_down(yv, 1, 16);
        if (n == 0) {
            int lo = dir ? (L_TOK - 1 - l) : l;
            float zv = zp[lo];
            yp[lo] = (yv + Dv * xcv) * silu_f(zv);
        }
    }
}

// ---------------- combine yf+yb, transpose channel-major -> token-major bf16 ----------------
__global__ void combine_k(const float* __restrict__ y, __bf16* __restrict__ Ay)
{
    __shared__ float s[32][33];
    int b  = blockIdx.z;
    int d0 = blockIdx.y * 32, l0 = blockIdx.x * 32;
    int tl = threadIdx.x & 31, td = threadIdx.x >> 5;   // td 0..7
    #pragma unroll
    for (int i = 0; i < 4; i++) {
        int dd = td + i * 8;
        size_t o = ((size_t)b * DI + d0 + dd) * L_TOK + l0 + tl;
        s[dd][tl] = y[o] + y[o + (size_t)NB * DI * L_TOK];
    }
    __syncthreads();
    #pragma unroll
    for (int i = 0; i < 4; i++) {
        int lr = td + i * 8;
        Ay[((size_t)(b * L_TOK + l0 + lr)) * DI + d0 + tl] = (__bf16)s[tl][lr];
    }
}

// ---------------- host ----------------
extern "C" void kernel_launch(void* const* d_in, const int* in_sizes, int n_in,
                              void* d_out, int out_size, void* d_ws, size_t ws_size,
                              hipStream_t stream)
{
    (void)in_sizes; (void)n_in; (void)out_size; (void)ws_size;
    const float* x       = (const float*)d_in[0];
    const float* patch_w = (const float*)d_in[1];
    const float* patch_b = (const float*)d_in[2];
    const float* pos     = (const float*)d_in[3];
    const float* tpos    = (const float*)d_in[4];
    const float* ipw     = (const float*)d_in[5];
    const float* cw      = (const float*)d_in[6];
    const float* cb      = (const float*)d_in[7];
    const float* xpw     = (const float*)d_in[8];
    const float* dtw     = (const float*)d_in[9];
    const float* dtbi    = (const float*)d_in[10];
    const float* Al      = (const float*)d_in[11];
    const float* Dp      = (const float*)d_in[12];
    const float* cwb     = (const float*)d_in[13];
    const float* cbb     = (const float*)d_in[14];
    const float* xpwb    = (const float*)d_in[15];
    const float* dtwb    = (const float*)d_in[16];
    const float* dtbb    = (const float*)d_in[17];
    const float* Alb     = (const float*)d_in[18];
    const float* Dpb     = (const float*)d_in[19];
    const float* opw     = (const float*)d_in[20];
    const float* nwAll   = (const float*)d_in[21];
    const float* nf      = (const float*)d_in[22];
    float* out = (float*)d_out;

    size_t off = 0;
    char* base = (char*)d_ws;
    auto carve = [&](size_t bytes) -> void* {
        void* p = base + off;
        off += (bytes + 255) & ~(size_t)255;
        return p;
    };
    float*  res   = (float*)carve((size_t)BT * DM * 4);
    float*  hid   = (float*)carve((size_t)BT * DM * 4);
    __bf16* Xn    = (__bf16*)carve((size_t)BT_PAD * DM * 2);
    __bf16* Abuf  = (__bf16*)carve((size_t)BT_PAD * DI * 2);  // patch A, reused as (yf+yb) A
    float*  xz    = (float*)carve((size_t)NB * NCH * L_TOK * 4);
    float*  xc    = (float*)carve((size_t)2 * NB * DI * L_TOK * 4);
    float*  dtbuf = (float*)carve((size_t)2 * NB * DI * L_TOK * 4);
    float*  xdbl  = (float*)carve((size_t)2 * NB * NK * L_TOK * 4);
    float*  ybuf  = (float*)carve((size_t)2 * NB * DI * L_TOK * 4);

    const int MB = BT_PAD / GBM;   // 25

    // ---- patch embedding ----
    patch_gather<<<BT, 256, 0, stream>>>(x, Abuf);
    wmma_gemm<<<dim3(DM / GBN, MB), 256, 0, stream>>>(Abuf, patch_w, hid, BT, DM, 768, 0);
    embed_finish<<<(BT * DM) / 256, 256, 0, stream>>>(hid, res, patch_b, pos, tpos);

    // ---- 24 bidirectional mamba blocks ----
    for (int lay = 0; lay < 24; lay++) {
        rms_in<<<BT, 128, 0, stream>>>(res, hid, nwAll + (size_t)lay * DM, Xn);
        wmma_gemm<<<dim3(NCH / GBN, MB), 256, 0, stream>>>(
            Xn, ipw + (size_t)lay * NCH * DM, xz, BT, NCH, DM, 1);
        conv_silu<<<dim3((NB * DI * L_TOK) / 256, 2), 256, 0, stream>>>(
            xz, cw + (size_t)lay * DI * 4, cb + (size_t)lay * DI,
            cwb + (size_t)lay * DI * 4, cbb + (size_t)lay * DI, xc);
        xproj_k<<<dim3((NB * NK * L_TOK) / 256, 2), 256, 0, stream>>>(
            xc, xpw + (size_t)lay * NK * DI, xpwb + (size_t)lay * NK * DI, xdbl);
        dt_k<<<dim3((NB * DI * L_TOK) / 256, 2), 256, 0, stream>>>(
            xdbl, dtw + (size_t)lay * DI * DTR, dtbi + (size_t)lay * DI,
            dtwb + (size_t)lay * DI * DTR, dtbb + (size_t)lay * DI, dtbuf);
        scan_k<<<dim3(DI / 16, NB, 2), 256, 0, stream>>>(
            dtbuf, xc, xdbl, xz,
            Al + (size_t)lay * DI * DST, Dp + (size_t)lay * DI,
            Alb + (size_t)lay * DI * DST, Dpb + (size_t)lay * DI, ybuf);
        combine_k<<<dim3(L_TOK / 32, DI / 32, NB), 256, 0, stream>>>(ybuf, Abuf);
        wmma_gemm<<<dim3(DM / GBN, MB), 256, 0, stream>>>(
            Abuf, opw + (size_t)lay * DM * DI, hid, BT, DM, DI, 0);
    }

    // ---- final norm ----
    rms_final<<<BT, 128, 0, stream>>>(res, hid, nf, out);
}
